// RWKVB_layers_31825707663875
// MI455X (gfx1250) — compile-verified, matrix operand order checked by reading, not run on hardware
//
#include <hip/hip_runtime.h>
#include <hip/hip_bf16.h>

// ---------------------------------------------------------------------------
// Vision-RWKV (B=8, CIN=4096, H=W=32, C=256, L=4, HID=1024) for gfx1250.
// All GEMMs: v_wmma_f32_16x16x32_bf16, 64x32 macro-tile per wave
// (8 WMMA per 32-K step; compiler-scheduled loads/clauses — explicit
// double-buffering caused spills and was reverted).
// ---------------------------------------------------------------------------

typedef __attribute__((ext_vector_type(16))) __bf16 v16bf;
typedef __attribute__((ext_vector_type(8)))  float  v8f;

#define B_DIM   8
#define CIN_DIM 4096
#define HI      32
#define WIW     32
#define HW      1024
#define C_DIM   256
#define T_DIM   1024
#define M_DIM   8192          // B * H * W tokens
#define HID_DIM 1024
#define L_DIM   4

// Load one 16-lane-striped bf16 fragment row: 8 contiguous bf16 at p, plus
// 8 contiguous bf16 at p+16 (second K-chunk), per the CDNA5 16-bit A layout.
__device__ __forceinline__ v16bf load_frag(const __bf16* p) {
  v16bf r;
  const uint4* q = reinterpret_cast<const uint4*>(p);
  uint4 c0 = q[0];          // K chunk 0 (8 bf16 = 16 bytes)
  uint4 c1 = q[2];          // K chunk 1 at +16 elements = +32 bytes
  reinterpret_cast<uint4*>(&r)[0] = c0;
  reinterpret_cast<uint4*>(&r)[1] = c1;
  return r;
}

// ---------------------------------------------------------------------------
// BN folding: scale = g * rsqrt(var+eps), bias = b - mean*scale
// ---------------------------------------------------------------------------
__global__ void bn_prep_k(const float* __restrict__ g, const float* __restrict__ b,
                          const float* __restrict__ mean, const float* __restrict__ var,
                          float* __restrict__ scale, float* __restrict__ bias) {
  int c = threadIdx.x;
  float s = g[c] * rsqrtf(var[c] + 1e-5f);
  scale[c] = s;
  bias[c]  = b[c] - mean[c] * s;
}

// ---------------------------------------------------------------------------
// Transpose+convert weight: W[K,N] f32 -> Bt[N,K] bf16
// ---------------------------------------------------------------------------
__global__ void tw_k(const float* __restrict__ W, __bf16* __restrict__ Bt,
                     int Kd, int Nd) {
  int idx = blockIdx.x * blockDim.x + threadIdx.x;
  if (idx >= Kd * Nd) return;
  int n = idx / Kd, k = idx - n * Kd;
  Bt[idx] = (__bf16)W[(size_t)k * Nd + n];
}

// ---------------------------------------------------------------------------
// x[B,CIN,H,W] f32 -> xT[M=B*HW, CIN] bf16 via LDS-tiled transpose.
// ---------------------------------------------------------------------------
__global__ void xcvt_k(const float* __restrict__ x, __bf16* __restrict__ xT) {
  __shared__ float tile[32][33];
  int b  = blockIdx.x >> 12;                 // 8 * 128 * 32 blocks
  int k0 = ((blockIdx.x >> 5) & 127) * 32;   // cin tile
  int w0 = (blockIdx.x & 31) * 32;           // hw tile
  int tr = threadIdx.x >> 5;                 // 0..7
  int tc = threadIdx.x & 31;                 // 0..31
#pragma unroll
  for (int i = 0; i < 4; ++i) {
    int kk = tr + i * 8;
    tile[kk][tc] = x[((size_t)b * CIN_DIM + k0 + kk) * HW + w0 + tc];
  }
  __syncthreads();
#pragma unroll
  for (int i = 0; i < 4; ++i) {
    int ww = tr + i * 8;
    xT[((size_t)b * HW + w0 + ww) * CIN_DIM + k0 + tc] = (__bf16)tile[tc][ww];
  }
}

// ---------------------------------------------------------------------------
// GEMM: C[M,N] f32 = A[M,K] bf16 (row-major) x Bt[N,K] bf16.  M = 8192.
// Wave computes a 64x32 tile: per 32-K step, 4 A-frag + 2 B-frag loads
// feed 8 v_wmma_f32_16x16x32_bf16.  Optional fused BN+ReLU epilogue.
// ---------------------------------------------------------------------------
template <bool EPI>
__global__ void gemm_t(const __bf16* __restrict__ A, const __bf16* __restrict__ Bt,
                       float* __restrict__ Cm,
                       const float* __restrict__ scale, const float* __restrict__ bias,
                       int N, int K) {
  int wid  = (blockIdx.x * blockDim.x + threadIdx.x) >> 5;
  int lane = threadIdx.x & 31;
  int ntiles = N >> 5;                       // 32-wide n-tiles
  int mblk = wid / ntiles;
  int nt   = wid - mblk * ntiles;
  int m0 = mblk * 64, n0 = nt * 32;
  int l15 = lane & 15, hh = lane >> 4;

  const __bf16* bp0 = Bt + (size_t)(n0 + l15) * K + hh * 8;
  const __bf16* bp1 = bp0 + (size_t)16 * K;
  const __bf16* ap  = A + (size_t)(m0 + l15) * K + hh * 8;
  size_t rstep = (size_t)16 * K;

  v8f acc[8];
#pragma unroll
  for (int t = 0; t < 8; ++t)
#pragma unroll
    for (int j = 0; j < 8; ++j) acc[t][j] = 0.f;

  for (int k0 = 0; k0 < K; k0 += 32) {
    __builtin_prefetch(ap + k0 + 256, 0, 1);      // global_prefetch_b8
    __builtin_prefetch(bp0 + k0 + 256, 0, 1);
    v16bf b0 = load_frag(bp0 + k0);
    v16bf b1 = load_frag(bp1 + k0);
#pragma unroll
    for (int t = 0; t < 4; ++t) {
      v16bf af = load_frag(ap + (size_t)t * rstep + k0);
      acc[t] = __builtin_amdgcn_wmma_f32_16x16x32_bf16(
          false, af, false, b0, (short)0, acc[t], false, false);
      acc[4 + t] = __builtin_amdgcn_wmma_f32_16x16x32_bf16(
          false, af, false, b1, (short)0, acc[4 + t], false, false);
    }
  }

  int col0 = n0 + l15;
  int col1 = n0 + 16 + l15;
  float s0 = 0.f, b0e = 0.f, s1 = 0.f, b1e = 0.f;
  if (EPI) { s0 = scale[col0]; b0e = bias[col0]; s1 = scale[col1]; b1e = bias[col1]; }
#pragma unroll
  for (int t = 0; t < 4; ++t)
#pragma unroll
    for (int j = 0; j < 8; ++j) {
      size_t r = (size_t)(m0 + t * 16 + hh * 8 + j) * N;
      if (EPI) {
        Cm[r + col0] = fmaxf(acc[t][j] * s0 + b0e, 0.f);
        Cm[r + col1] = fmaxf(acc[4 + t][j] * s1 + b1e, 0.f);
      } else {
        Cm[r + col0] = acc[t][j];
        Cm[r + col1] = acc[4 + t][j];
      }
    }
}

// ---------------------------------------------------------------------------
// LayerNorm over C=256, one wave per row; lane handles 8 strided channels.
// ---------------------------------------------------------------------------
__global__ void ln_k(const float* __restrict__ in, float* __restrict__ out,
                     const float* __restrict__ g, const float* __restrict__ b) {
  int row  = (blockIdx.x * blockDim.x + threadIdx.x) >> 5;
  int lane = threadIdx.x & 31;
  const float* r = in + (size_t)row * C_DIM;
  float v[8];
  float s = 0.f;
#pragma unroll
  for (int j = 0; j < 8; ++j) { v[j] = r[lane + j * 32]; s += v[j]; }
#pragma unroll
  for (int o = 16; o > 0; o >>= 1) s += __shfl_xor(s, o, 32);
  float mu = s * (1.f / 256.f);
  float var = 0.f;
#pragma unroll
  for (int j = 0; j < 8; ++j) { float d = v[j] - mu; var += d * d; }
#pragma unroll
  for (int o = 16; o > 0; o >>= 1) var += __shfl_xor(var, o, 32);
  float inv = rsqrtf(var * (1.f / 256.f) + 1e-6f);
  float* w = out + (size_t)row * C_DIM;
#pragma unroll
  for (int j = 0; j < 8; ++j) {
    int c = lane + j * 32;
    w[c] = (v[j] - mu) * inv * g[c] + b[c];
  }
}

// ---------------------------------------------------------------------------
// q_shift + token/channel mixing -> up to 3 bf16 A buffers.
// Channel quarter selects neighbor: 0:w-1, 1:w+1, 2:h-1, 3:h+1 (zero pad).
// ---------------------------------------------------------------------------
__global__ void qmix_k(const float* __restrict__ ln,
                       const float* __restrict__ m0, const float* __restrict__ m1,
                       const float* __restrict__ m2,
                       __bf16* __restrict__ o0, __bf16* __restrict__ o1,
                       __bf16* __restrict__ o2) {
  int idx = blockIdx.x * blockDim.x + threadIdx.x;
  int c = idx & 255;
  int m = idx >> 8;
  int w = m & 31, h = (m >> 5) & 31;
  float xv = ln[idx];
  float xx = 0.f;
  switch (c >> 6) {
    case 0:  if (w > 0)  xx = ln[idx - C_DIM];        break;
    case 1:  if (w < 31) xx = ln[idx + C_DIM];        break;
    case 2:  if (h > 0)  xx = ln[idx - WIW * C_DIM];  break;
    default: if (h < 31) xx = ln[idx + WIW * C_DIM];  break;
  }
  float a0 = m0[c];
  o0[idx] = (__bf16)(xv * a0 + xx * (1.f - a0));
  if (o1) { float a1 = m1[c]; o1[idx] = (__bf16)(xv * a1 + xx * (1.f - a1)); }
  if (o2) { float a2 = m2[c]; o2[idx] = (__bf16)(xv * a2 + xx * (1.f - a2)); }
}

// ---------------------------------------------------------------------------
// Numerically-stable RWKV WKV scan: one thread per (b,c), T=1024 steps.
// ---------------------------------------------------------------------------
__global__ void wkv_k(const float* __restrict__ decay, const float* __restrict__ first,
                      const float* __restrict__ k, const float* __restrict__ v,
                      float* __restrict__ out) {
  int tid = blockIdx.x * blockDim.x + threadIdx.x;   // 0..2047
  int b = tid >> 8, c = tid & 255;
  float w = -__expf(decay[c] * (1.f / (float)T_DIM));
  float u = first[c] * (1.f / (float)T_DIM);
  float aa = 0.f, bb = 0.f, pp = -1e38f;
  size_t base = (size_t)b * T_DIM * C_DIM + c;
  for (int t = 0; t < T_DIM; ++t) {
    size_t ix = base + (size_t)t * C_DIM;
    float kt = k[ix], vt = v[ix];
    float ww = u + kt;
    float q  = fmaxf(pp, ww);
    float e1 = __expf(pp - q), e2 = __expf(ww - q);
    out[ix] = (e1 * aa + e2 * vt) / (e1 * bb + e2);
    float ww2 = pp + w;
    float q2  = fmaxf(ww2, kt);
    float f1 = __expf(ww2 - q2), f2 = __expf(kt - q2);
    aa = f1 * aa + f2 * vt;
    bb = f1 * bb + f2;
    pp = q2;
  }
}

// sigmoid(r) * wkv -> bf16 A buffer (feeds Wo GEMM)
__global__ void gate_k(const float* __restrict__ r, const float* __restrict__ wv,
                       __bf16* __restrict__ o) {
  int idx = blockIdx.x * blockDim.x + threadIdx.x;
  float s = 1.f / (1.f + __expf(-r[idx]));
  o[idx] = (__bf16)(s * wv[idx]);
}

// y += ls[c] * g
__global__ void resadd_k(float* __restrict__ y, const float* __restrict__ g,
                         const float* __restrict__ ls) {
  int idx = blockIdx.x * blockDim.x + threadIdx.x;
  y[idx] += ls[idx & 255] * g[idx];
}

// hid -> bf16(relu(hid)^2)
__global__ void relusq_k(const float* __restrict__ h, __bf16* __restrict__ o) {
  int idx = blockIdx.x * blockDim.x + threadIdx.x;
  float v = fmaxf(h[idx], 0.f);
  o[idx] = (__bf16)(v * v);
}

// y += ls[c] * sigmoid(r) * kv
__global__ void cmres_k(float* __restrict__ y, const float* __restrict__ kv,
                        const float* __restrict__ r, const float* __restrict__ ls) {
  int idx = blockIdx.x * blockDim.x + threadIdx.x;
  float s = 1.f / (1.f + __expf(-r[idx]));
  y[idx] += ls[idx & 255] * s * kv[idx];
}

// ---------------------------------------------------------------------------
extern "C" void kernel_launch(void* const* d_in, const int* in_sizes, int n_in,
                              void* d_out, int out_size, void* d_ws, size_t ws_size,
                              hipStream_t stream) {
  const float* x       = (const float*)d_in[0];
  const float* conv_w  = (const float*)d_in[1];
  const float* bn_g    = (const float*)d_in[2];
  const float* bn_b    = (const float*)d_in[3];
  const float* bn_mean = (const float*)d_in[4];
  const float* bn_var  = (const float*)d_in[5];
  const float* ln1_g   = (const float*)d_in[6];
  const float* ln1_b   = (const float*)d_in[7];
  const float* ln2_g   = (const float*)d_in[8];
  const float* ln2_b   = (const float*)d_in[9];
  const float* ls1     = (const float*)d_in[10];
  const float* ls2     = (const float*)d_in[11];
  const float* amix_k  = (const float*)d_in[12];
  const float* amix_v  = (const float*)d_in[13];
  const float* amix_r  = (const float*)d_in[14];
  const float* a_decay = (const float*)d_in[15];
  const float* a_first = (const float*)d_in[16];
  const float* aWk     = (const float*)d_in[17];
  const float* aWv     = (const float*)d_in[18];
  const float* aWr     = (const float*)d_in[19];
  const float* aWo     = (const float*)d_in[20];
  const float* mmix_k  = (const float*)d_in[21];
  const float* mmix_r  = (const float*)d_in[22];
  const float* mWk     = (const float*)d_in[23];
  const float* mWv     = (const float*)d_in[24];
  const float* mWr     = (const float*)d_in[25];
  const float* norm_g  = (const float*)d_in[26];
  const float* norm_b  = (const float*)d_in[27];
  float* out = (float*)d_out;
  char* ws = (char*)d_ws;

  constexpr size_t MB = 1ull << 20;
  // Workspace layout (phase-based reuse, ~85 MB peak):
  //   xT (8..72 MB) lives only during the conv; the same region is reused by
  //   all per-layer buffers afterwards.
  float*  y    = (float*)(ws + 0);          // 8 MB residual stream [8192,256]
  __bf16* xT   = (__bf16*)(ws + 8 * MB);    // 64 MB bf16 [8192,4096] (conv only)
  float*  lnb  = (float*)(ws + 8 * MB);     // 8 MB LN output
  __bf16* A0   = (__bf16*)(ws + 16 * MB);   // 4 MB bf16 [8192,256]
  __bf16* A1   = (__bf16*)(ws + 20 * MB);   // 4 MB
  __bf16* A2   = (__bf16*)(ws + 24 * MB);   // 4 MB
  float*  G0   = (float*)(ws + 28 * MB);    // 8 MB (k / later kv)
  float*  G1   = (float*)(ws + 36 * MB);    // 8 MB (v / later r)
  float*  G2   = (float*)(ws + 44 * MB);    // 8 MB (r spatial)
  float*  wkvb = (float*)(ws + 52 * MB);    // 8 MB
  float*  hidf = (float*)(ws + 28 * MB);    // 32 MB [8192,1024] (reuses G0..wkv)
  __bf16* hidb = (__bf16*)(ws + 60 * MB);   // 16 MB bf16 [8192,1024]
  float*  kvb  = (float*)(ws + 28 * MB);    // 8 MB (after hidf dead)
  float*  rb   = (float*)(ws + 36 * MB);    // 8 MB

  size_t p = 76 * MB;
  __bf16* convT = (__bf16*)(ws + p); p += (size_t)CIN_DIM * C_DIM * 2;   // 2 MB
  __bf16 *aWkT[L_DIM], *aWvT[L_DIM], *aWrT[L_DIM], *aWoT[L_DIM];
  __bf16 *mWkT[L_DIM], *mWvT[L_DIM], *mWrT[L_DIM];
  for (int i = 0; i < L_DIM; ++i) {
    aWkT[i] = (__bf16*)(ws + p); p += 256 * 256 * 2;
    aWvT[i] = (__bf16*)(ws + p); p += 256 * 256 * 2;
    aWrT[i] = (__bf16*)(ws + p); p += 256 * 256 * 2;
    aWoT[i] = (__bf16*)(ws + p); p += 256 * 256 * 2;
    mWkT[i] = (__bf16*)(ws + p); p += 1024 * 256 * 2;  // Bt[1024,256]
    mWvT[i] = (__bf16*)(ws + p); p += 256 * 1024 * 2;  // Bt[256,1024]
    mWrT[i] = (__bf16*)(ws + p); p += 256 * 256 * 2;
  }
  float* bnscale = (float*)(ws + p); p += 256 * 4;
  float* bnbias  = (float*)(ws + p); p += 256 * 4;

  // ---- prep: fold BN, transpose+convert weights, convert x to bf16 [M,K] ----
  bn_prep_k<<<1, 256, 0, stream>>>(bn_g, bn_b, bn_mean, bn_var, bnscale, bnbias);
  tw_k<<<(CIN_DIM * C_DIM) / 256, 256, 0, stream>>>(conv_w, convT, CIN_DIM, C_DIM);
  for (int i = 0; i < L_DIM; ++i) {
    tw_k<<<256, 256, 0, stream>>>(aWk + (size_t)i * 65536, aWkT[i], 256, 256);
    tw_k<<<256, 256, 0, stream>>>(aWv + (size_t)i * 65536, aWvT[i], 256, 256);
    tw_k<<<256, 256, 0, stream>>>(aWr + (size_t)i * 65536, aWrT[i], 256, 256);
    tw_k<<<256, 256, 0, stream>>>(aWo + (size_t)i * 65536, aWoT[i], 256, 256);
    tw_k<<<1024, 256, 0, stream>>>(mWk + (size_t)i * 262144, mWkT[i], 256, 1024);
    tw_k<<<1024, 256, 0, stream>>>(mWv + (size_t)i * 262144, mWvT[i], 1024, 256);
    tw_k<<<256, 256, 0, stream>>>(mWr + (size_t)i * 65536, mWrT[i], 256, 256);
  }
  xcvt_k<<<32768, 256, 0, stream>>>(x, xT);   // 8 * 128 * 32 tiles

  // ---- 1x1 conv + BN + ReLU -> y [8192,256] ----
  // waves = (8192/64)*(256/32) = 1024 -> 128 blocks of 8 waves
  gemm_t<true><<<128, 256, 0, stream>>>(xT, convT, y, bnscale, bnbias,
                                        C_DIM, CIN_DIM);

  const int gElem = M_DIM;        // 8192 blocks of 256 for [8192,256] elementwise
  const int gG256  = 128;         // (128 * 8)  / 8 wave-blocks, N=256
  const int gG1024 = 512;         // (128 * 32) / 8 wave-blocks, N=1024

  for (int i = 0; i < L_DIM; ++i) {
    size_t oC = (size_t)i * C_DIM;
    // -------- spatial mix --------
    ln_k<<<1024, 256, 0, stream>>>(y, lnb, ln1_g + oC, ln1_b + oC);
    qmix_k<<<gElem, 256, 0, stream>>>(lnb, amix_k + oC, amix_v + oC, amix_r + oC,
                                      A0, A1, A2);
    gemm_t<false><<<gG256, 256, 0, stream>>>(A0, aWkT[i], G0, nullptr, nullptr, 256, 256);
    gemm_t<false><<<gG256, 256, 0, stream>>>(A1, aWvT[i], G1, nullptr, nullptr, 256, 256);
    gemm_t<false><<<gG256, 256, 0, stream>>>(A2, aWrT[i], G2, nullptr, nullptr, 256, 256);
    wkv_k<<<8, 256, 0, stream>>>(a_decay + oC, a_first + oC, G0, G1, wkvb);
    gate_k<<<gElem, 256, 0, stream>>>(G2, wkvb, A0);                  // sigmoid(r)*wkv
    gemm_t<false><<<gG256, 256, 0, stream>>>(A0, aWoT[i], G0, nullptr, nullptr, 256, 256);
    resadd_k<<<gElem, 256, 0, stream>>>(y, G0, ls1 + oC);
    // -------- channel mix --------
    ln_k<<<1024, 256, 0, stream>>>(y, lnb, ln2_g + oC, ln2_b + oC);
    qmix_k<<<gElem, 256, 0, stream>>>(lnb, mmix_k + oC, mmix_r + oC, nullptr,
                                      A0, A1, nullptr);
    gemm_t<false><<<gG1024, 256, 0, stream>>>(A0, mWkT[i], hidf, nullptr, nullptr, 1024, 256);
    relusq_k<<<32768, 256, 0, stream>>>(hidf, hidb);                  // relu(k)^2
    gemm_t<false><<<gG256, 256, 0, stream>>>(hidb, mWvT[i], kvb, nullptr, nullptr, 256, 1024);
    gemm_t<false><<<gG256, 256, 0, stream>>>(A1, mWrT[i], rb, nullptr, nullptr, 256, 256);
    cmres_k<<<gElem, 256, 0, stream>>>(y, kvb, rb, ls2 + oC);
  }

  // ---- final layer norm -> d_out ----
  ln_k<<<1024, 256, 0, stream>>>(y, out, norm_g, norm_b);
}
